// QueryLayer_6854767804923
// MI455X (gfx1250) — matrix operand order repaired and checked by reading, not compiled, not run on hardware
//
#include <hip/hip_runtime.h>

typedef _Float16 half8 __attribute__((ext_vector_type(8)));
typedef _Float16 v16h  __attribute__((ext_vector_type(16)));
typedef float    v8f   __attribute__((ext_vector_type(8)));
typedef float    f4    __attribute__((ext_vector_type(4)));

#define N_NODES 4096
#define IN_F    256
#define OUT_F   64
#define ALPHA   0.2f
#define NUM_REL 4

// ---------------------------------------------------------------------------
// Kernel 1: h = input @ W  (4096x256 @ 256x64) via v_wmma_f32_16x16x32_f16.
// One wave per 16x16 output tile. Writes h (f32, row major) and h^T (f16).
// ---------------------------------------------------------------------------
__global__ __launch_bounds__(32)
void k_h_gemm(const float* __restrict__ inp, const float* __restrict__ W,
              float* __restrict__ h32, _Float16* __restrict__ ht) {
  const int m0   = blockIdx.x * 16;
  const int n0   = blockIdx.y * 16;
  const int lane = threadIdx.x;
  const int l15  = lane & 15;
  const bool hi  = lane >= 16;

  const float* arow = inp + (size_t)(m0 + l15) * IN_F;
  v8f acc = {};

  for (int k0 = 0; k0 < IN_F; k0 += 32) {
    // A operand: f16 16x32. Lane holds M=l15; two 8-run K groups.
    v16h A;
    {
      const int ka = k0 + (hi ? 8 : 0);
      f4 r0 = *(const f4*)(arow + ka);
      f4 r1 = *(const f4*)(arow + ka + 4);
      f4 r2 = *(const f4*)(arow + ka + 16);
      f4 r3 = *(const f4*)(arow + ka + 20);
#pragma unroll
      for (int i = 0; i < 4; ++i) {
        A[i]      = (_Float16)r0[i];
        A[4 + i]  = (_Float16)r1[i];
        A[8 + i]  = (_Float16)r2[i];
        A[12 + i] = (_Float16)r3[i];
      }
    }
    // B operand: f16 32x16. Lane holds N=l15; two 8-run K groups.
    v16h B;
    {
      const int kb = k0 + (hi ? 8 : 0);
#pragma unroll
      for (int i = 0; i < 16; ++i)
        B[i] = (_Float16)W[(size_t)(kb + (i < 8 ? i : i + 8)) * OUT_F + n0 + l15];
    }
    acc = __builtin_amdgcn_wmma_f32_16x16x32_f16(false, A, false, B,
                                                 (short)0, acc, false, false);
  }

#pragma unroll
  for (int k = 0; k < 8; ++k) {
    const int row = m0 + k + (hi ? 8 : 0);
    const float v = acc[k];
    h32[(size_t)row * OUT_F + n0 + l15] = v;
    ht[(size_t)(n0 + l15) * N_NODES + row] = (_Float16)v;  // h^T for B loads
  }
}

// ---------------------------------------------------------------------------
// Kernel 2: s1 = h @ a[:64], s2 = h @ a[64:]. Tiny.
// ---------------------------------------------------------------------------
__global__ __launch_bounds__(64)
void k_s(const float* __restrict__ h32, const float* __restrict__ a,
         float* __restrict__ s1, float* __restrict__ s2) {
  const int i = blockIdx.x * 64 + threadIdx.x;
  float x1 = 0.f, x2 = 0.f;
#pragma unroll 8
  for (int f = 0; f < OUT_F; ++f) {
    const float hv = h32[(size_t)i * OUT_F + f];
    x1 += hv * a[f];
    x2 += hv * a[OUT_F + f];
  }
  s1[i] = x1;
  s2[i] = x2;
}

// ---------------------------------------------------------------------------
// Kernel 3: per-row online softmax stats over att = adj * leaky_relu(s1+s2^T).
// One block per row; online (m, S) with rescale, then LDS tree reduce.
// ---------------------------------------------------------------------------
__global__ __launch_bounds__(256)
void k_stats(const float* __restrict__ adj, const float* __restrict__ s1,
             const float* __restrict__ s2, float* __restrict__ mrow,
             float* __restrict__ invS) {
  __shared__ float sm[256];
  __shared__ float ss[256];
  const int i = blockIdx.x;
  const int t = threadIdx.x;
  const float s1i = s1[i];
  const float* arow = adj + (size_t)i * N_NODES;

  float m = -1e30f, S = 0.f;
  for (int j = t; j < N_NODES; j += 256) {
    float e = s1i + s2[j];
    e = e > 0.f ? e : ALPHA * e;
    const float att = arow[j] * e;
    if (att > m) {
      S = S * __expf(m - att) + 1.f;
      m = att;
    } else {
      S += __expf(att - m);
    }
  }
  sm[t] = m; ss[t] = S;
  __syncthreads();
  for (int w = 128; w > 0; w >>= 1) {
    if (t < w) {
      const float m2 = sm[t + w], S2 = ss[t + w];
      const float M = fmaxf(sm[t], m2);
      ss[t] = ss[t] * __expf(sm[t] - M) + S2 * __expf(m2 - M);
      sm[t] = M;
    }
    __syncthreads();
  }
  if (t == 0) {
    mrow[i] = sm[0];
    invS[i] = 1.f / ss[0];
  }
}

// ---------------------------------------------------------------------------
// Kernel 4: h' = softmax(att) @ h, out = elu(h') replicated NUM_REL times.
// Block = 16 rows x 128 threads (4 waves, wave w owns features [16w,16w+16)).
// Per 128-wide K step: waves cooperatively build P tile (16x128 f16) in LDS,
// then each wave does 4 WMMAs (A from LDS via b128 loads, B from h^T).
// ---------------------------------------------------------------------------
#define P_STRIDE 136  // 128 + 8 halves padding -> conflict-free ds_load_b128

__global__ __launch_bounds__(128)
void k_attn(const float* __restrict__ adj, const _Float16* __restrict__ ht,
            const float* __restrict__ s1, const float* __restrict__ s2,
            const float* __restrict__ mrow, const float* __restrict__ invS,
            float* __restrict__ out) {
  __shared__ _Float16 Pl[16 * P_STRIDE];

  const int m0   = blockIdx.x * 16;
  const int wave = threadIdx.x >> 5;
  const int lane = threadIdx.x & 31;
  const int l15  = lane & 15;
  const bool hi  = lane >= 16;
  const int n0   = wave * 16;

  // P-generation assignment: this lane produces row l15, 16 K values.
  const int   pko  = hi ? 16 : 0;
  const float s1v  = s1[m0 + l15];
  const float mv   = mrow[m0 + l15];
  const float* arow = adj + (size_t)(m0 + l15) * N_NODES;
  const _Float16* bbase = ht + (size_t)(n0 + l15) * N_NODES;

  v8f acc = {};

  for (int k0 = 0; k0 < N_NODES; k0 += 128) {
    const int kc = k0 + wave * 32 + pko;
    if (k0 + 128 < N_NODES)
      __builtin_prefetch(arow + kc + 128, 0, 1);  // global_prefetch next chunk

    // --- build 16 unnormalized softmax values -> LDS ---
    half8 pa, pb;
    {
      f4 a0 = *(const f4*)(arow + kc);
      f4 a1 = *(const f4*)(arow + kc + 4);
      f4 a2 = *(const f4*)(arow + kc + 8);
      f4 a3 = *(const f4*)(arow + kc + 12);
      f4 t0 = *(const f4*)(s2 + kc);
      f4 t1 = *(const f4*)(s2 + kc + 4);
      f4 t2 = *(const f4*)(s2 + kc + 8);
      f4 t3 = *(const f4*)(s2 + kc + 12);
#pragma unroll
      for (int q = 0; q < 4; ++q) {
        float e0 = s1v + t0[q]; e0 = e0 > 0.f ? e0 : ALPHA * e0;
        float e1 = s1v + t1[q]; e1 = e1 > 0.f ? e1 : ALPHA * e1;
        float e2 = s1v + t2[q]; e2 = e2 > 0.f ? e2 : ALPHA * e2;
        float e3 = s1v + t3[q]; e3 = e3 > 0.f ? e3 : ALPHA * e3;
        pa[q]     = (_Float16)__expf(a0[q] * e0 - mv);
        pa[4 + q] = (_Float16)__expf(a1[q] * e1 - mv);
        pb[q]     = (_Float16)__expf(a2[q] * e2 - mv);
        pb[4 + q] = (_Float16)__expf(a3[q] * e3 - mv);
      }
    }
    {
      _Float16* dst = &Pl[l15 * P_STRIDE + wave * 32 + pko];
      *(half8*)dst = pa;
      *(half8*)(dst + 8) = pb;
    }
    __syncthreads();

    // --- 4 WMMAs over the 128-wide P tile ---
#pragma unroll
    for (int c = 0; c < 4; ++c) {
      const _Float16* ap = &Pl[l15 * P_STRIDE + c * 32 + (hi ? 8 : 0)];
      half8 a0 = *(const half8*)ap;
      half8 a1 = *(const half8*)(ap + 16);
      v16h A = __builtin_shufflevector(a0, a1, 0, 1, 2, 3, 4, 5, 6, 7,
                                       8, 9, 10, 11, 12, 13, 14, 15);
      const _Float16* bp = bbase + k0 + c * 32 + (hi ? 8 : 0);
      half8 b0 = *(const half8*)bp;
      half8 b1 = *(const half8*)(bp + 16);
      v16h B = __builtin_shufflevector(b0, b1, 0, 1, 2, 3, 4, 5, 6, 7,
                                       8, 9, 10, 11, 12, 13, 14, 15);
      acc = __builtin_amdgcn_wmma_f32_16x16x32_f16(false, A, false, B,
                                                   (short)0, acc, false, false);
    }
    __syncthreads();
  }

  // --- finalize: normalize, ELU, replicate over relations ---
#pragma unroll
  for (int k = 0; k < 8; ++k) {
    const int row = m0 + k + (hi ? 8 : 0);
    float v = acc[k] * invS[row];
    v = v > 0.f ? v : (__expf(v) - 1.f);
    const int col = n0 + l15;
#pragma unroll
    for (int r = 0; r < NUM_REL; ++r)
      out[(size_t)r * N_NODES * OUT_F + (size_t)row * OUT_F + col] = v;
  }
}

// ---------------------------------------------------------------------------
extern "C" void kernel_launch(void* const* d_in, const int* in_sizes, int n_in,
                              void* d_out, int out_size, void* d_ws, size_t ws_size,
                              hipStream_t stream) {
  (void)in_sizes; (void)n_in; (void)out_size; (void)ws_size;
  const float* inp = (const float*)d_in[0];
  const float* adj = (const float*)d_in[1];
  const float* W   = (const float*)d_in[2];
  const float* a   = (const float*)d_in[3];
  float* out = (float*)d_out;

  char* ws = (char*)d_ws;
  float*    h32  = (float*)ws;                                    // 1 MB
  _Float16* ht   = (_Float16*)(ws + (1 << 20));                   // 512 KB
  float*    s1   = (float*)(ws + (1 << 20) + (512 << 10));        // 16 KB
  float*    s2   = s1 + N_NODES;                                  // 16 KB
  float*    mrow = s2 + N_NODES;                                  // 16 KB
  float*    invS = mrow + N_NODES;                                // 16 KB

  k_h_gemm<<<dim3(N_NODES / 16, OUT_F / 16), 32, 0, stream>>>(inp, W, h32, ht);
  k_s<<<N_NODES / 64, 64, 0, stream>>>(h32, a, s1, s2);
  k_stats<<<N_NODES, 256, 0, stream>>>(adj, s1, s2, mrow, invS);
  k_attn<<<N_NODES / 16, 128, 0, stream>>>(adj, ht, s1, s2, mrow, invS, out);
}
